// TransformerBlock_82660940579281
// MI455X (gfx1250) — matrix operand order repaired and checked by reading, not compiled, not run on hardware
//
#include <hip/hip_runtime.h>

typedef _Float16 h16;
typedef __attribute__((ext_vector_type(8)))  _Float16 v8h;
typedef __attribute__((ext_vector_type(16))) _Float16 v16h;
typedef __attribute__((ext_vector_type(8)))  float    v8f;
typedef __attribute__((ext_vector_type(2)))  float    v2f;

constexpr int B = 4, N = 2048, D = 256, K = 16;
constexpr int NPTS = B * N;
constexpr int WV = 4;          // waves (points) per block in pt_kernel

// ---------------- kernel 0: f32 -> f16 weight conversion ----------------
__global__ void cvt_kernel(const float* __restrict__ src, h16* __restrict__ dst, int n) {
  int i = blockIdx.x * blockDim.x + threadIdx.x;
  if (i < n) dst[i] = (h16)src[i];
}

// ---------------- kernel 1: kNN (one wave32 per query point) ----------------
__global__ __launch_bounds__(256) void knn_kernel(const float* __restrict__ xyz,
                                                  int* __restrict__ knn) {
  __shared__ float sd[8][N];                       // 64 KB
  const int wave = threadIdx.x >> 5;
  const int lane = threadIdx.x & 31;
  const int p = blockIdx.x * 8 + wave;             // global point id
  const int b = p >> 11;                           // N == 2048
  const float qx = xyz[p * 3 + 0];
  const float qy = xyz[p * 3 + 1];
  const float qz = xyz[p * 3 + 2];
  const float* xb = xyz + (size_t)b * N * 3;
  for (int m = lane; m < N; m += 32) {
    float dx = qx - xb[m * 3 + 0];
    float dy = qy - xb[m * 3 + 1];
    float dz = qz - xb[m * 3 + 2];
    sd[wave][m] = dx * dx + dy * dy + dz * dz;
  }
  __syncthreads();
  for (int pass = 0; pass < K; ++pass) {
    float best = 3.4e38f;
    int   bi   = 0x7fffffff;
    for (int m = lane; m < N; m += 32) {
      float v = sd[wave][m];
      if (v < best) { best = v; bi = m; }
    }
    // wave32 argmin reduction, ties -> lowest index (matches top_k order)
    for (int off = 16; off >= 1; off >>= 1) {
      float ob = __shfl_xor(best, off);
      int   oi = __shfl_xor(bi, off);
      if (ob < best || (ob == best && oi < bi)) { best = ob; bi = oi; }
    }
    if (lane == 0) {
      knn[p * K + pass] = bi;
      sd[wave][bi] = 3.4e38f;      // remove winner for next pass
    }
    __syncthreads();
  }
}

// ---------------- WMMA helper: one 16x16 output tile of A(16xD) @ W(DxD) ----------------
// Abuf: shared-mem f16 row base for this lane's M-row (lane L -> M = L&15).
// A layout (16-bit A, 16x32): lanes 0-15 hold K {0..7,16..23}, lanes 16-31 hold K {8..15,24..31}.
// W: global f16, row-major [Din][Dout]; B layout: lane -> N = L&15, K order same as A.
__device__ inline v8f wmma_tile(const h16* Abuf, const h16* __restrict__ W,
                                int n0, int lane) {
  const int hi = lane >> 4;
  v8f c = {};
#pragma unroll
  for (int ks = 0; ks < D / 32; ++ks) {
    const int k0 = ks * 32;
    v8h lo = *(const v8h*)(Abuf + k0 + 8 * hi);
    v8h hh = *(const v8h*)(Abuf + k0 + 16 + 8 * hi);
    v16h a = __builtin_shufflevector(lo, hh, 0, 1, 2, 3, 4, 5, 6, 7,
                                     8, 9, 10, 11, 12, 13, 14, 15);
    v16h bm;
#pragma unroll
    for (int s = 0; s < 8; ++s) {
      bm[s]     = W[(size_t)(k0 + 8 * hi + s) * D + n0 + (lane & 15)];
      bm[s + 8] = W[(size_t)(k0 + 16 + 8 * hi + s) * D + n0 + (lane & 15)];
    }
    c = __builtin_amdgcn_wmma_f32_16x16x32_f16(false, a, false, bm,
                                               (short)0, c, false, false);
  }
  return c;
}

// ---------------- kernel 2: fused point-transformer block (one wave per point) ----------
__global__ __launch_bounds__(32 * WV) void pt_kernel(
    const float* __restrict__ xyz, const float* __restrict__ feat,
    const float* __restrict__ wd1, const float* __restrict__ bd1,
    const float* __restrict__ bd2, const float* __restrict__ bg1,
    const float* __restrict__ bg2,
    const h16* __restrict__ wd2h, const h16* __restrict__ wg1h,
    const h16* __restrict__ wg2h,
    const int* __restrict__ knn, float* __restrict__ out) {
  __shared__ __align__(32) h16 sXK[WV][K * D];   // x_k   (16 x 256) f16, 8 KB/wave
  __shared__ __align__(32) h16 sTMP[WV][K * D];  // h1/g1 staging,     8 KB/wave
  const int wave = threadIdx.x >> 5;
  const int lane = threadIdx.x & 31;
  const int p  = blockIdx.x * WV + wave;
  const int b  = p >> 11;
  const int m  = lane & 15;       // neighbor row owned by this lane (A rows / D cols)
  const int hi = lane >> 4;
  h16* XK  = sXK[wave];
  h16* TMP = sTMP[wave];

  // Each lane holds neighbor index of row m (both 16-lane halves replicate).
  const int nid = knn[p * K + m];

  // delta for neighbor m
  const float qx = xyz[p * 3 + 0], qy = xyz[p * 3 + 1], qz = xyz[p * 3 + 2];
  const float* xb = xyz + (size_t)b * N * 3;
  const float d0 = qx - xb[nid * 3 + 0];
  const float d1 = qy - xb[nid * 3 + 1];
  const float d2 = qz - xb[nid * 3 + 2];

  // ---- layer d1 (3 -> 256) via fp32 WMMA 16x16x4, K padded 3 -> 4 with zeros.
  // A (16x4 f32, 2 VGPRs): lanes 0-15 hold {K0,K1}, lanes 16-31 hold {K2,K3}.
  // Both lane halves already hold this row's delta, so A is just registers:
  v2f a1;
  a1.x = hi ? d2 : d0;
  a1.y = hi ? 0.0f : d1;
  for (int nt = 0; nt < D / 16; ++nt) {
    const int n0 = nt * 16;
    // B (4x16 f32, 2 VGPRs): lane = N column; lanes 0-15 hold {K0,K1}, 16-31 {K2,K3}.
    v2f b1;
    b1.x = hi ? wd1[2 * D + n0 + m] : wd1[0 * D + n0 + m];
    b1.y = hi ? 0.0f : wd1[1 * D + n0 + m];
    v8f c = {};
    c = __builtin_amdgcn_wmma_f32_16x16x4_f32(false, a1, false, b1,
                                              (short)0, c, false, false);
    const int nc = n0 + m;
    const float bias = bd1[nc];
#pragma unroll
    for (int r = 0; r < 8; ++r)
      TMP[(r + 8 * hi) * D + nc] = (h16)fmaxf(c[r] + bias, 0.0f);
  }

  const float* fb = feat + (size_t)b * N * D;

  // ---- layer d2 (WMMA) + gathered features:  x_k = h1 @ wd2 + bd2 + feat[knn]
  for (int nt = 0; nt < D / 16; ++nt) {
    const int n0 = nt * 16;
    v8f c = wmma_tile(TMP + m * D, wd2h, n0, lane);
    const int nc = n0 + m;                 // output column owned by this lane
    const float bias = bd2[nc];
#pragma unroll
    for (int r = 0; r < 8; ++r) {
      const int row = r + 8 * hi;          // C/D layout: VGPR r holds M = r + 8*hi
      const int rid = __shfl(nid, row);    // neighbor index for that row
      const float fv = fb[(size_t)rid * D + nc];
      XK[row * D + nc] = (h16)(c[r] + bias + fv);
    }
  }

  // ---- layer g1 (WMMA): g1 = relu(x_k @ wg1 + bg1)
  for (int nt = 0; nt < D / 16; ++nt) {
    const int n0 = nt * 16;
    v8f c = wmma_tile(XK + m * D, wg1h, n0, lane);
    const int nc = n0 + m;
    const float bias = bg1[nc];
#pragma unroll
    for (int r = 0; r < 8; ++r) {
      const int row = r + 8 * hi;
      TMP[row * D + nc] = (h16)fmaxf(c[r] + bias, 0.0f);
    }
  }

  // ---- layer g2 (WMMA) + softmax over the 16 neighbors + weighted sum + residual
  for (int nt = 0; nt < D / 16; ++nt) {
    const int n0 = nt * 16;
    v8f c = wmma_tile(TMP + m * D, wg2h, n0, lane);
    const int nc = n0 + m;
    const float bias = bg2[nc];
    float l[8];
    float mx = -3.4e38f;
#pragma unroll
    for (int r = 0; r < 8; ++r) { l[r] = c[r] + bias; mx = fmaxf(mx, l[r]); }
    mx = fmaxf(mx, __shfl_xor(mx, 16));          // 16 rows live in lane pair L, L^16
    float s = 0.0f;
#pragma unroll
    for (int r = 0; r < 8; ++r) { l[r] = __expf(l[r] - mx); s += l[r]; }
    s += __shfl_xor(s, 16);
    const float inv = 1.0f / s;
    float acc = 0.0f;
#pragma unroll
    for (int r = 0; r < 8; ++r)
      acc += l[r] * (float)XK[(r + 8 * hi) * D + nc];
    acc *= inv;
    acc += __shfl_xor(acc, 16);
    if (hi == 0)
      out[(size_t)p * D + nc] = feat[(size_t)p * D + nc] + acc;  // residual add
  }
}

extern "C" void kernel_launch(void* const* d_in, const int* in_sizes, int n_in,
                              void* d_out, int out_size, void* d_ws, size_t ws_size,
                              hipStream_t stream) {
  const float* xyz  = (const float*)d_in[0];
  const float* feat = (const float*)d_in[1];
  const float* wd1  = (const float*)d_in[2];
  const float* bd1  = (const float*)d_in[3];
  const float* wd2  = (const float*)d_in[4];
  const float* bd2  = (const float*)d_in[5];
  const float* wg1  = (const float*)d_in[6];
  const float* bg1  = (const float*)d_in[7];
  const float* wg2  = (const float*)d_in[8];
  const float* bg2  = (const float*)d_in[9];
  float* out = (float*)d_out;

  char* ws = (char*)d_ws;
  int* knn  = (int*)ws;                                  // 512 KB
  h16* wd2h = (h16*)(ws + (size_t)NPTS * K * sizeof(int));
  h16* wg1h = wd2h + (size_t)D * D;                      // +128 KB
  h16* wg2h = wg1h + (size_t)D * D;                      // +128 KB

  cvt_kernel<<<(D * D + 255) / 256, 256, 0, stream>>>(wd2, wd2h, D * D);
  cvt_kernel<<<(D * D + 255) / 256, 256, 0, stream>>>(wg1, wg1h, D * D);
  cvt_kernel<<<(D * D + 255) / 256, 256, 0, stream>>>(wg2, wg2h, D * D);

  knn_kernel<<<NPTS / 8, 256, 0, stream>>>(xyz, knn);

  pt_kernel<<<NPTS / WV, 32 * WV, 0, stream>>>(xyz, feat, wd1, bd1, bd2, bg1, bg2,
                                               wd2h, wg1h, wg2h, knn, out);
}
